// Y_decoder_5583457485496
// MI455X (gfx1250) — matrix operand order repaired and checked by reading, not compiled, not run on hardware
//
#include <hip/hip_runtime.h>
#include <hip/hip_bf16.h>

#define NN 50000
#define NE 400000
#define KDIM 192     // 64 + 128
#define HID 512

typedef __attribute__((ext_vector_type(16))) __bf16 v16bf;
typedef __attribute__((ext_vector_type(8)))  float  v8f;

union ABfrag { v16bf v; float4 f4[2]; };

__device__ __forceinline__ void atomAddF32(float* p, float v) {
    __hip_atomic_fetch_add(p, v, __ATOMIC_RELAXED, __HIP_MEMORY_SCOPE_AGENT);
}

// ---------------- degree / normalization ----------------
__global__ void k_deg_init(float* deg) {
    int i = blockIdx.x * blockDim.x + threadIdx.x;
    if (i < NN) deg[i] = 1.0f;   // self-loop contributes 1
}

__global__ void k_deg_edges(const int* __restrict__ eidx, float* deg) {
    int e = blockIdx.x * blockDim.x + threadIdx.x;
    if (e < NE) atomAddF32(&deg[eidx[NE + e]], 1.0f);
}

__global__ void k_dinv(const float* __restrict__ deg, float* __restrict__ dinv) {
    int i = blockIdx.x * blockDim.x + threadIdx.x;
    if (i < NN) dinv[i] = rsqrtf(deg[i]);   // deg >= 1 always
}

// ---------------- bf16 conversion of latent = [u_Y | X] ----------------
__global__ void k_cvt_latent(const float* __restrict__ X, const float* __restrict__ uY,
                             __bf16* __restrict__ A1) {
    int tid = blockIdx.x * blockDim.x + threadIdx.x;
    if (tid >= NN * KDIM) return;
    int i = tid / KDIM;
    int f = tid - i * KDIM;
    float v = (f < 64) ? uY[i * 64 + f] : X[i * 128 + (f - 64)];
    A1[tid] = (__bf16)v;
}

// ---------------- repack W1 into WMMA B-fragment layout ----------------
// Bfrag[kc][ct][lane][j]  (6 x 32 x 32 x 16 bf16)
// lane 0-15  : col = ct*16 + lane,      j -> K = kc*32 + j
// lane 16-31 : col = ct*16 + (lane-16), j -> K = kc*32 + 16 + j
__global__ void k_pack_w1(const float* __restrict__ W1, __bf16* __restrict__ Bf) {
    int tid = blockIdx.x * blockDim.x + threadIdx.x;
    if (tid >= 6 * 32 * 32 * 16) return;
    int j    =  tid        & 15;
    int lane = (tid >> 4)  & 31;
    int ct   = (tid >> 9)  & 31;
    int kc   =  tid >> 14;
    int k = kc * 32 + (lane >> 4) * 16 + j;
    int c = ct * 16 + (lane & 15);
    Bf[tid] = (__bf16)W1[k * HID + c];
}

// ---------------- GEMM1: XW1 = latent @ W1 (bf16 WMMA, f32 acc) ----------
// grid(3125, 4), block 256 (8 waves). wave w -> col-tile blockIdx.y*8+w.
// Epilogue also seeds H with the self-loop message dinv[i]^2 * xw.
__global__ void __launch_bounds__(256)
k_gemm1(const __bf16* __restrict__ A1, const __bf16* __restrict__ Bf,
        const float* __restrict__ dinv,
        float* __restrict__ XW1, float* __restrict__ H) {
    const int wave = threadIdx.x >> 5;
    const int lane = threadIdx.x & 31;
    const int hf   = lane >> 4;
    const int l16  = lane & 15;
    const int rowTile = blockIdx.x * 16;
    const int ct      = blockIdx.y * 8 + wave;      // 0..31
    const int m       = rowTile + l16;              // A row for this lane

    const __bf16* arow = A1 + (size_t)m * KDIM;
    v8f acc = {};
#pragma unroll
    for (int kc = 0; kc < 6; ++kc) {
        const int kb = kc * 32;
        ABfrag fa, fb;
        fa.f4[0] = *reinterpret_cast<const float4*>(arow + kb + hf * 16);
        fa.f4[1] = *reinterpret_cast<const float4*>(arow + kb + hf * 16 + 8);
        const float4* pb = reinterpret_cast<const float4*>(
            Bf + (((size_t)kc * 32 + ct) * 32 + lane) * 16);
        fb.f4[0] = pb[0];
        fb.f4[1] = pb[1];
        acc = __builtin_amdgcn_wmma_f32_16x16x32_bf16(
            false, fa.v, false, fb.v, (short)0, acc, false, false);
    }
    const int colBase = ct * 16 + l16;
#pragma unroll
    for (int i = 0; i < 8; ++i) {
        int row = rowTile + hf * 8 + i;
        float d = dinv[row];
        size_t off = (size_t)row * HID + colBase;
        float v = acc[i];
        XW1[off] = v;
        H[off]   = d * d * v;   // self-loop term
    }
}

// ---------------- layer-1 edge scatter: H[c] += dinv[r]*dinv[c]*XW1[r] ----
// one wave per edge, coalesced 128B row chunks
__global__ void __launch_bounds__(256)
k_scatter1(const int* __restrict__ eidx, const float* __restrict__ dinv,
           const float* __restrict__ XW1, float* __restrict__ H) {
    const int wave = threadIdx.x >> 5;
    const int lane = threadIdx.x & 31;
    const int e = blockIdx.x * 8 + wave;
    if (e >= NE) return;
    const int r = eidx[e];
    const int c = eidx[NE + e];
    const float nrm = dinv[r] * dinv[c];
    const float* __restrict__ src = XW1 + (size_t)r * HID;
    float* __restrict__ dst = H + (size_t)c * HID;
#pragma unroll
    for (int f = lane; f < HID; f += 32)
        atomAddF32(&dst[f], nrm * src[f]);
}

// ---------------- fused bias + ReLU + layer-2 GEMV (+ self-loop init of Y) -
// one wave per node; h_relu is never materialized
__global__ void __launch_bounds__(256)
k_gemv2(const float* __restrict__ H, const float* __restrict__ b1,
        const float* __restrict__ W2, const float* __restrict__ dinv,
        float* __restrict__ XW2, float* __restrict__ Y2) {
    const int wave = threadIdx.x >> 5;
    const int lane = threadIdx.x & 31;
    const int i = blockIdx.x * 8 + wave;
    if (i >= NN) return;
    const float* __restrict__ hrow = H + (size_t)i * HID;
    float a0 = 0.f, a1 = 0.f;
#pragma unroll
    for (int f = lane; f < HID; f += 32) {
        float h = hrow[f] + b1[f];
        h = h > 0.f ? h : 0.f;
        a0 += h * W2[2 * f];
        a1 += h * W2[2 * f + 1];
    }
#pragma unroll
    for (int off = 16; off > 0; off >>= 1) {
        a0 += __shfl_down(a0, off, 32);
        a1 += __shfl_down(a1, off, 32);
    }
    if (lane == 0) {
        float s = dinv[i] * dinv[i];
        XW2[2 * i]     = a0;
        XW2[2 * i + 1] = a1;
        Y2[2 * i]      = s * a0;   // self-loop term
        Y2[2 * i + 1]  = s * a1;
    }
}

// ---------------- layer-2 edge scatter (2 floats/edge) ----------------
__global__ void k_scatter2(const int* __restrict__ eidx, const float* __restrict__ dinv,
                           const float* __restrict__ XW2, float* __restrict__ Y2) {
    int e = blockIdx.x * blockDim.x + threadIdx.x;
    if (e >= NE) return;
    int r = eidx[e];
    int c = eidx[NE + e];
    float nrm = dinv[r] * dinv[c];
    atomAddF32(&Y2[2 * c],     nrm * XW2[2 * r]);
    atomAddF32(&Y2[2 * c + 1], nrm * XW2[2 * r + 1]);
}

// ---------------- bias + softmax over 2 classes ----------------
__global__ void k_softmax(const float* __restrict__ Y2, const float* __restrict__ b2,
                          float* __restrict__ out) {
    int i = blockIdx.x * blockDim.x + threadIdx.x;
    if (i >= NN) return;
    float y0 = Y2[2 * i] + b2[0];
    float y1 = Y2[2 * i + 1] + b2[1];
    float m = fmaxf(y0, y1);
    float e0 = __expf(y0 - m);
    float e1 = __expf(y1 - m);
    float inv = 1.f / (e0 + e1);
    out[2 * i]     = e0 * inv;
    out[2 * i + 1] = e1 * inv;
}

extern "C" void kernel_launch(void* const* d_in, const int* in_sizes, int n_in,
                              void* d_out, int out_size, void* d_ws, size_t ws_size,
                              hipStream_t stream) {
    const int*   eidx = (const int*)  d_in[0];   // [2, 400000]
    const float* X    = (const float*)d_in[1];   // [50000, 128]
    const float* uY   = (const float*)d_in[2];   // [50000, 64]
    const float* W1   = (const float*)d_in[3];   // [192, 512]
    const float* b1   = (const float*)d_in[4];   // [512]
    const float* W2   = (const float*)d_in[5];   // [512, 2]
    const float* b2   = (const float*)d_in[6];   // [2]
    float* out = (float*)d_out;                  // [50000, 2]

    // workspace layout (all 16B aligned by construction)
    float*  deg  = (float*)d_ws;                          // 50000
    float*  dinv = deg + NN;                              // 50000
    __bf16* A1   = (__bf16*)(dinv + NN);                  // 50000*192 bf16
    __bf16* Bf   = A1 + (size_t)NN * KDIM;                // 6*32*32*16 bf16
    float*  XW1  = (float*)(Bf + 6 * 32 * 32 * 16);       // 50000*512
    float*  H    = XW1 + (size_t)NN * HID;                // 50000*512
    float*  XW2  = H + (size_t)NN * HID;                  // 50000*2
    float*  Y2   = XW2 + 2 * NN;                          // 50000*2

    const int T = 256;
    k_deg_init  <<<(NN + T - 1) / T, T, 0, stream>>>(deg);
    k_deg_edges <<<(NE + T - 1) / T, T, 0, stream>>>(eidx, deg);
    k_dinv      <<<(NN + T - 1) / T, T, 0, stream>>>(deg, dinv);
    k_cvt_latent<<<(NN * KDIM + T - 1) / T, T, 0, stream>>>(X, uY, A1);
    k_pack_w1   <<<(6 * 32 * 32 * 16 + T - 1) / T, T, 0, stream>>>(W1, Bf);

    dim3 g1(NN / 16, HID / (16 * 8));   // (3125, 4)
    k_gemm1     <<<g1, T, 0, stream>>>(A1, Bf, dinv, XW1, H);

    k_scatter1  <<<NE / 8, T, 0, stream>>>(eidx, dinv, XW1, H);
    k_gemv2     <<<(NN + 7) / 8, T, 0, stream>>>(H, b1, W2, dinv, XW2, Y2);
    k_scatter2  <<<(NE + T - 1) / T, T, 0, stream>>>(eidx, dinv, XW2, Y2);
    k_softmax   <<<(NN + T - 1) / T, T, 0, stream>>>(Y2, b2, out);
}
